// PatchCore_5128190952110
// MI455X (gfx1250) — compile-verified, matrix operand order checked by reading, not run on hardware
//
#include <hip/hip_runtime.h>
#include <hip/hip_bf16.h>

// ---------------- problem constants (from reference) ----------------
#define QROWS   6272      // 8 * 28 * 28
#define NBANK   50000
#define DDIM    1536
#define BATCH   8
#define PGRID   28
#define OUT_HW  224

// ---------------- GEMM tiling ----------------
#define MT 128            // queries per workgroup tile
#define NT 128            // bank rows per workgroup tile
#define KC 32             // K chunk (halves) staged per buffer
#define LDS_STRIDE 40     // 32 + 8 halves pad -> 80B row stride, 16B aligned, conflict-free

// gfx1250 async global->LDS path (guarded: falls back to sync staging if absent)
#if defined(__has_builtin)
#if __has_builtin(__builtin_amdgcn_global_load_async_to_lds_b128) && \
    __has_builtin(__builtin_amdgcn_s_wait_asynccnt)
#define USE_ASYNC_LDS 1
#endif
#endif

typedef __attribute__((ext_vector_type(16))) __bf16 v16bf;
typedef __attribute__((ext_vector_type(8)))  float  v8f;
typedef __attribute__((ext_vector_type(4)))  int    v4i;

union FragBF {
    v16bf v;
    uint2 q[4];
};

// ---------------- helpers ----------------
static __device__ __forceinline__ unsigned short f32_to_bf16_rne(float f) {
    unsigned int u = __float_as_uint(f);
    u = (u + 0x7FFFu + ((u >> 16) & 1u)) >> 16;
    return (unsigned short)u;
}
static __device__ __forceinline__ float bf16_bits_to_f32(unsigned short h) {
    return __uint_as_float(((unsigned int)h) << 16);
}

// ---------------- kernel 1: init running-min buffer to +inf ----------------
__global__ void pc_init_min(unsigned int* __restrict__ pm, int n) {
    int i = blockIdx.x * blockDim.x + threadIdx.x;
    if (i < n) pm[i] = 0x7F800000u;   // +inf
}

// ---------------- kernel 2: fused fp32 -> (bf16 hi, bf16 lo) split + row sqnorm ----
__global__ __launch_bounds__(256) void pc_preprocess(const float* __restrict__ x,
                                                     unsigned short* __restrict__ hi,
                                                     unsigned short* __restrict__ lo,
                                                     float* __restrict__ norms,
                                                     int rows) {
    int wid  = (blockIdx.x * 256 + threadIdx.x) >> 5;
    int lane = threadIdx.x & 31;
    if (wid >= rows) return;
    const float4* p = (const float4*)(x + (size_t)wid * DDIM);
    size_t hbase = (size_t)wid * DDIM;
    float s = 0.0f;
#pragma unroll
    for (int j = 0; j < 12; ++j) {
        int idx = lane + 32 * j;               // float4 index within row
        float4 v = p[idx];
        s += v.x * v.x + v.y * v.y + v.z * v.z + v.w * v.w;
        unsigned short h0 = f32_to_bf16_rne(v.x);
        unsigned short h1 = f32_to_bf16_rne(v.y);
        unsigned short h2 = f32_to_bf16_rne(v.z);
        unsigned short h3 = f32_to_bf16_rne(v.w);
        unsigned short l0 = f32_to_bf16_rne(v.x - bf16_bits_to_f32(h0));
        unsigned short l1 = f32_to_bf16_rne(v.y - bf16_bits_to_f32(h1));
        unsigned short l2 = f32_to_bf16_rne(v.z - bf16_bits_to_f32(h2));
        unsigned short l3 = f32_to_bf16_rne(v.w - bf16_bits_to_f32(h3));
        uint2 hv, lv;
        hv.x = (unsigned)h0 | ((unsigned)h1 << 16);
        hv.y = (unsigned)h2 | ((unsigned)h3 << 16);
        lv.x = (unsigned)l0 | ((unsigned)l1 << 16);
        lv.y = (unsigned)l2 | ((unsigned)l3 << 16);
        *(uint2*)(hi + hbase + (size_t)idx * 4) = hv;
        *(uint2*)(lo + hbase + (size_t)idx * 4) = lv;
    }
#pragma unroll
    for (int m = 16; m >= 1; m >>= 1) s += __shfl_xor(s, m, 32);
    if (lane == 0) norms[wid] = s;
}

// ---------------- LDS staging: 128 rows x 32 halves per plane ------------------
// Each of 256 threads moves two 16B segments per plane. Rows past rowClamp are
// clamped to a valid row (finite garbage; results discarded by epilogue guard).
static __device__ __forceinline__ void stage_plane(const unsigned short* __restrict__ g,
                                                   unsigned short* __restrict__ s,
                                                   int rowBase, int rowClamp, int kc, int tid) {
#pragma unroll
    for (int i = 0; i < 2; ++i) {
        int idx = tid + (i << 8);          // 0..511
        int row = idx >> 2;                // 0..127
        int seg = idx & 3;                 // 4 x 8-half segments per row
        int grow = rowBase + row;
        if (grow > rowClamp) grow = rowClamp;
        const unsigned short* gp = g + (size_t)grow * DDIM + kc + seg * 8;
        unsigned short* sp = s + row * LDS_STRIDE + seg * 8;
#ifdef USE_ASYNC_LDS
        __builtin_amdgcn_global_load_async_to_lds_b128(
            (__attribute__((address_space(1))) v4i*)gp,
            (__attribute__((address_space(3))) v4i*)sp, 0, 0);
#else
        uint4 v = *(const uint4*)gp;
        uint2 a, b;
        a.x = v.x; a.y = v.y; b.x = v.z; b.y = v.w;
        *(uint2*)sp       = a;
        *(uint2*)(sp + 4) = b;
#endif
    }
}

static __device__ __forceinline__ void wait_staged_keep8() {
#ifdef USE_ASYNC_LDS
    __builtin_amdgcn_s_wait_asynccnt(8);   // 8 in-flight = the just-issued next chunk
#endif
}
static __device__ __forceinline__ void wait_staged_all() {
#ifdef USE_ASYNC_LDS
    __builtin_amdgcn_s_wait_asynccnt(0);
#endif
}

// 16x32 bf16 fragment from LDS (A and B^T share the same per-lane layout)
static __device__ __forceinline__ void load_frag(const unsigned short* __restrict__ s,
                                                 int row0, int lane, FragBF& f) {
    int r  = row0 + (lane & 15);
    int b0 = (lane & 16) ? 8 : 0;
    const unsigned short* p = s + r * LDS_STRIDE + b0;
    f.q[0] = *(const uint2*)(p);
    f.q[1] = *(const uint2*)(p + 4);
    f.q[2] = *(const uint2*)(p + 16);
    f.q[3] = *(const uint2*)(p + 20);
}

// ---------------- kernel 3: fused GEMM + min-distance --------------------------
// C = Qhi*Bhi + Qhi*Blo + Qlo*Bhi  (~fp32 dot);  d2 = qn + bn - 2C; global min.
__global__ __launch_bounds__(256) void pc_gemm_min(const unsigned short* __restrict__ Ahi,
                                                   const unsigned short* __restrict__ Alo,
                                                   const unsigned short* __restrict__ Bhi,
                                                   const unsigned short* __restrict__ Blo,
                                                   const float* __restrict__ qn,
                                                   const float* __restrict__ bn,
                                                   unsigned int* __restrict__ patch_min,
                                                   int nrows) {
    __shared__ __align__(16) unsigned short sAhi[2][MT * LDS_STRIDE];
    __shared__ __align__(16) unsigned short sAlo[2][MT * LDS_STRIDE];
    __shared__ __align__(16) unsigned short sBhi[2][NT * LDS_STRIDE];
    __shared__ __align__(16) unsigned short sBlo[2][NT * LDS_STRIDE];

    const int m0   = blockIdx.x * MT;
    const int n0   = blockIdx.y * NT;
    const int tid  = threadIdx.x;
    const int wave = tid >> 5;
    const int lane = tid & 31;
    const int wm   = wave & 1;   // 0..1 -> 64-row half
    const int wn   = wave >> 1;  // 0..3 -> 32-col quarter

    v8f acc[4][2];
#pragma unroll
    for (int i = 0; i < 4; ++i)
#pragma unroll
        for (int j = 0; j < 2; ++j) acc[i][j] = 0.0f;

    const int nchunks = DDIM / KC;   // 48

    // prologue: stage chunk 0 into buffer 0
    stage_plane(Ahi, sAhi[0], m0, QROWS - 1, 0, tid);
    stage_plane(Alo, sAlo[0], m0, QROWS - 1, 0, tid);
    stage_plane(Bhi, sBhi[0], n0, nrows - 1, 0, tid);
    stage_plane(Blo, sBlo[0], n0, nrows - 1, 0, tid);

    for (int c = 0; c < nchunks; ++c) {
        const int cur = c & 1;
        if (c + 1 < nchunks) {
            const int nxt = cur ^ 1;
            const int kc  = (c + 1) * KC;
            stage_plane(Ahi, sAhi[nxt], m0, QROWS - 1, kc, tid);
            stage_plane(Alo, sAlo[nxt], m0, QROWS - 1, kc, tid);
            stage_plane(Bhi, sBhi[nxt], n0, nrows - 1, kc, tid);
            stage_plane(Blo, sBlo[nxt], n0, nrows - 1, kc, tid);
            wait_staged_keep8();      // chunk c's 8 per-wave copies complete
        } else {
            wait_staged_all();
        }
        __syncthreads();              // chunk c visible to all waves

        FragBF bh[2], bl[2];
#pragma unroll
        for (int tn = 0; tn < 2; ++tn) {
            load_frag(sBhi[cur], 32 * wn + 16 * tn, lane, bh[tn]);
            load_frag(sBlo[cur], 32 * wn + 16 * tn, lane, bl[tn]);
        }
#pragma unroll
        for (int tm = 0; tm < 4; ++tm) {
            FragBF ah, al;
            load_frag(sAhi[cur], 64 * wm + 16 * tm, lane, ah);
            load_frag(sAlo[cur], 64 * wm + 16 * tm, lane, al);
#pragma unroll
            for (int tn = 0; tn < 2; ++tn) {
                acc[tm][tn] = __builtin_amdgcn_wmma_f32_16x16x32_bf16(
                    false, ah.v, false, bh[tn].v, (short)0, acc[tm][tn], false, false);
                acc[tm][tn] = __builtin_amdgcn_wmma_f32_16x16x32_bf16(
                    false, ah.v, false, bl[tn].v, (short)0, acc[tm][tn], false, false);
                acc[tm][tn] = __builtin_amdgcn_wmma_f32_16x16x32_bf16(
                    false, al.v, false, bh[tn].v, (short)0, acc[tm][tn], false, false);
            }
        }
        __syncthreads();              // all waves done reading buf[cur] before reuse
    }

    // epilogue: d2 = qn + bn - 2*dot ; min over this block's N columns, atomic-fold
    const int halfsel = (lane >> 4) * 8;     // lanes 16..31 hold M = r + 8
    const int ncol    = lane & 15;
#pragma unroll
    for (int tm = 0; tm < 4; ++tm) {
        int mbase = m0 + 64 * wm + 16 * tm + halfsel;
#pragma unroll
        for (int r = 0; r < 8; ++r) {
            int m = mbase + r;
            float qv = qn[m];
            float best = __uint_as_float(0x7F800000u);
#pragma unroll
            for (int tn = 0; tn < 2; ++tn) {
                int n = n0 + 32 * wn + 16 * tn + ncol;
                if (n < nrows) {
                    float d = qv + bn[n] - 2.0f * acc[tm][tn][r];
                    best = fminf(best, d);
                }
            }
            best = fmaxf(best, 0.0f);        // exact d2 >= 0; keeps uint-min valid
#pragma unroll
            for (int s = 8; s >= 1; s >>= 1) best = fminf(best, __shfl_xor(best, s, 16));
            if (ncol == 0) atomicMin(&patch_min[m], __float_as_uint(best));
        }
    }
}

// ---------------- kernel 4: per-image max of 28x28 patch scores ---------------
__global__ void pc_image_scores(const unsigned int* __restrict__ pm, float* __restrict__ out) {
    __shared__ float red[8];
    int b = blockIdx.x, t = threadIdx.x;
    float mx = -3.402823466e+38f;
    for (int i = t; i < PGRID * PGRID; i += 256)
        mx = fmaxf(mx, __uint_as_float(pm[b * PGRID * PGRID + i]));
#pragma unroll
    for (int s = 16; s >= 1; s >>= 1) mx = fmaxf(mx, __shfl_xor(mx, s, 32));
    if ((t & 31) == 0) red[t >> 5] = mx;
    __syncthreads();
    if (t == 0) {
        float r = red[0];
#pragma unroll
        for (int w = 1; w < 8; ++w) r = fmaxf(r, red[w]);
        out[b] = r;
    }
}

// ---------------- kernel 5: bilinear 28x28 -> 224x224 (half-pixel centers) -----
__global__ void pc_resize(const unsigned int* __restrict__ pm, float* __restrict__ out) {
    int idx = blockIdx.x * blockDim.x + threadIdx.x;
    const int HW = OUT_HW * OUT_HW;
    if (idx >= BATCH * HW) return;
    int b = idx / HW;
    int rem = idx - b * HW;
    int y = rem / OUT_HW, x = rem - (rem / OUT_HW) * OUT_HW;
    float sy = fminf(fmaxf((y + 0.5f) * (float)PGRID / OUT_HW - 0.5f, 0.0f), (float)(PGRID - 1));
    float sx = fminf(fmaxf((x + 0.5f) * (float)PGRID / OUT_HW - 0.5f, 0.0f), (float)(PGRID - 1));
    int y0 = (int)sy, x0 = (int)sx;
    int y1 = min(y0 + 1, PGRID - 1), x1 = min(x0 + 1, PGRID - 1);
    float fy = sy - y0, fx = sx - x0;
    const unsigned int* p = pm + b * PGRID * PGRID;
    float v00 = __uint_as_float(p[y0 * PGRID + x0]);
    float v01 = __uint_as_float(p[y0 * PGRID + x1]);
    float v10 = __uint_as_float(p[y1 * PGRID + x0]);
    float v11 = __uint_as_float(p[y1 * PGRID + x1]);
    float top = v00 + (v01 - v00) * fx;
    float bot = v10 + (v11 - v10) * fx;
    out[BATCH + idx] = top + (bot - top) * fy;
}

// ---------------- launcher ----------------------------------------------------
extern "C" void kernel_launch(void* const* d_in, const int* in_sizes, int n_in,
                              void* d_out, int out_size, void* d_ws, size_t ws_size,
                              hipStream_t stream) {
    (void)in_sizes; (void)n_in; (void)out_size; (void)ws_size;
    const float* features = (const float*)d_in[0];   // [6272 x 1536]
    const float* bank     = (const float*)d_in[1];   // [50000 x 1536]
    float* out = (float*)d_out;                      // [8] scores + [8*224*224] masks

    // workspace layout (256B aligned sections)
    char* w = (char*)d_ws;
    size_t off = 0;
    auto take = [&](size_t bytes) { void* p = w + off; off += (bytes + 255) & ~(size_t)255; return p; };
    unsigned int*   patch_min = (unsigned int*)  take((size_t)QROWS * 4);
    float*          qn        = (float*)         take((size_t)QROWS * 4);
    float*          bnorm     = (float*)         take((size_t)NBANK * 4);
    unsigned short* feat_hi   = (unsigned short*)take((size_t)QROWS * DDIM * 2);
    unsigned short* feat_lo   = (unsigned short*)take((size_t)QROWS * DDIM * 2);
    unsigned short* bank_hi   = (unsigned short*)take((size_t)NBANK * DDIM * 2);
    unsigned short* bank_lo   = (unsigned short*)take((size_t)NBANK * DDIM * 2);

    // 1) running-min init
    pc_init_min<<<(QROWS + 255) / 256, 256, 0, stream>>>(patch_min, QROWS);

    // 2) fused split + norms (wave per row)
    pc_preprocess<<<(QROWS * 32 + 255) / 256, 256, 0, stream>>>(features, feat_hi, feat_lo, qn, QROWS);
    pc_preprocess<<<(NBANK * 32 + 255) / 256, 256, 0, stream>>>(bank, bank_hi, bank_lo, bnorm, NBANK);

    // 3) fused WMMA GEMM + min: grid x = M tiles (L2 reuse of bank slice), y = N tiles
    dim3 grid((QROWS + MT - 1) / MT, (NBANK + NT - 1) / NT);
    pc_gemm_min<<<grid, 256, 0, stream>>>(feat_hi, feat_lo, bank_hi, bank_lo,
                                          qn, bnorm, patch_min, NBANK);

    // 4) image scores (max over 784 patches per image)
    pc_image_scores<<<BATCH, 256, 0, stream>>>(patch_min, out);

    // 5) bilinear upsample masks
    pc_resize<<<(BATCH * OUT_HW * OUT_HW + 255) / 256, 256, 0, stream>>>(patch_min, out);
}
